// Matformer_28089086116286
// MI455X (gfx1250) — matrix operand order repaired and checked by reading, not compiled, not run on hardware
//
#include <hip/hip_runtime.h>
#include <hip/hip_bf16.h>

// ---------------------------------------------------------------------------
// Matformer forward for MI455X (gfx1250, wave32).
// Dense math via v_wmma_f32_16x16x32_bf16 (fp32 accumulate), bf16 activation
// storage. Weights are repacked once per launch into WMMA B-fragment order so
// the GEMM inner loop is pure contiguous 128-bit loads + WMMA. Wave tile is
// 16x32 (two accumulators share one A fragment). fp32 atomics for segment
// sums and BN statistics.
// ---------------------------------------------------------------------------

typedef __attribute__((ext_vector_type(16))) __bf16 v16bf;
typedef __attribute__((ext_vector_type(8)))  __bf16 bf8v;
typedef __attribute__((ext_vector_type(8)))  float  v8f;
typedef __attribute__((ext_vector_type(4)))  float  fx4;

#define ACT_NONE 0
#define ACT_SILU 1
#define ACT_SOFTPLUS 2

__device__ __forceinline__ float act_apply(float v, int act) {
    if (act == ACT_SILU)      return v / (1.f + __expf(-v));
    if (act == ACT_SOFTPLUS)  return (v > 20.f) ? v : __logf(1.f + __expf(v));
    return v;
}

// ----- A-fragment loaders: 16-bit A 16x32 layout (ISA 7.12.2) --------------
// lanes 0-15 : elems 0..7 -> K = 0..7,   elems 8..15 -> K = 16..23
// lanes 16-31: elems 0..7 -> K = 8..15,  elems 8..15 -> K = 24..31
__device__ __forceinline__ v16bf load_a_frag(const float* A, size_t row, int K,
                                             int kk, int hi) {
    const float* p = A + row * (size_t)K + kk + hi * 8;
    fx4 a0 = *(const fx4*)(p);
    fx4 a1 = *(const fx4*)(p + 4);
    fx4 a2 = *(const fx4*)(p + 16);
    fx4 a3 = *(const fx4*)(p + 20);
    v16bf a;
    a[0]=(__bf16)a0.x;  a[1]=(__bf16)a0.y;  a[2]=(__bf16)a0.z;  a[3]=(__bf16)a0.w;
    a[4]=(__bf16)a1.x;  a[5]=(__bf16)a1.y;  a[6]=(__bf16)a1.z;  a[7]=(__bf16)a1.w;
    a[8]=(__bf16)a2.x;  a[9]=(__bf16)a2.y;  a[10]=(__bf16)a2.z; a[11]=(__bf16)a2.w;
    a[12]=(__bf16)a3.x; a[13]=(__bf16)a3.y; a[14]=(__bf16)a3.z; a[15]=(__bf16)a3.w;
    return a;
}

__device__ __forceinline__ v16bf load_a_frag(const __bf16* A, size_t row, int K,
                                             int kk, int hi) {
    const __bf16* base = A + row * (size_t)K + kk + hi * 8;
    bf8v x = *(const bf8v*)(base);
    bf8v y = *(const bf8v*)(base + 16);
    v16bf a;
#pragma unroll
    for (int j = 0; j < 8; ++j) { a[j] = x[j]; a[8 + j] = y[j]; }
    return a;
}

// ----- Weight repack: W[K,N] fp32 -> bf16 WMMA B-fragment order ------------
// B 16-bit KxN striping (from ISA 7.12.4 dense-B layout):
//   lanes 0-15 : column = lane,    elems 0..7 -> K = kt*32 + 0..7,  elems 8..15 -> K = kt*32 + 16..23
//   lanes 16-31: column = lane-16, elems 0..7 -> K = kt*32 + 8..15, elems 8..15 -> K = kt*32 + 24..31
// Packed index: Wpk[ ((kt*(N/16) + nt)*32 + lane)*16 + j ]
// Rows k >= Ksrc are zero-filled (used to pad the K=92 embedding to 128).
__global__ void pack_w_kernel(const float* __restrict__ W,
                              __bf16* __restrict__ Wpk,
                              int Ksrc, int Kp, int N)
{
    int t = blockIdx.x * 256 + threadIdx.x;
    if (t >= Kp * N) return;
    int j    = t & 15;
    int l    = (t >> 4) & 31;
    int tile = t >> 9;
    int nT   = N >> 4;
    int nt   = tile % nT;
    int kt   = tile / nT;
    int col  = nt * 16 + (l & 15);
    int k    = kt * 32 + (l >> 4) * 8 + (j & 7) + (j >> 3) * 16;
    Wpk[t] = (k < Ksrc) ? (__bf16)W[(size_t)k * N + col] : (__bf16)0.f;
}

// ----- Generic WMMA GEMM: C = act(A @ W + bias) [* gate] -------------------
// 256 threads = 8 waves; block tile 64(M) x 64(N); wave tile 16x32 (two
// accumulators reuse one A fragment). Requires K % 32 == 0 and N % 64 == 0.
template <typename AT, typename CT>
__global__ __launch_bounds__(256) void wmma_gemm_kernel(
    const AT* __restrict__ A, const __bf16* __restrict__ Wpk,
    const float* __restrict__ bias, CT* __restrict__ C,
    const __bf16* __restrict__ gate, int M, int N, int K, int act)
{
    const int wave = threadIdx.x >> 5;
    const int lane = threadIdx.x & 31;
    const int lm   = lane & 15;
    const int hi   = lane >> 4;
    const int mTile = blockIdx.x * 64 + (wave >> 1) * 16;
    const int nBase = blockIdx.y * 64 + (wave & 1) * 32;
    const int nt0   = nBase >> 4;       // first of two 16-col B tiles
    const int nT    = N >> 4;

    // clamp A row: rows >= M duplicate row M-1; their outputs are masked at
    // the store, and control flow stays uniform (EXEC all-ones for WMMA).
    int rowA = mTile + lm;
    size_t rA = (rowA < M) ? (size_t)rowA : (size_t)(M - 1);

    v8f acc0 = {}, acc1 = {};
    int kt = 0;
    for (int kk = 0; kk < K; kk += 32, ++kt) {
        v16bf a  = load_a_frag(A, rA, K, kk, hi);
        const __bf16* bp =
            Wpk + (((size_t)kt * nT + nt0) * 32 + lane) * 16;
        v16bf b0 = *(const v16bf*)(bp);
        v16bf b1 = *(const v16bf*)(bp + 512);   // next 16-col tile (32*16)
        acc0 = __builtin_amdgcn_wmma_f32_16x16x32_bf16(
            false, a, false, b0, (short)0, acc0, false, false);
        acc1 = __builtin_amdgcn_wmma_f32_16x16x32_bf16(
            false, a, false, b1, (short)0, acc1, false, false);
    }

    const int col0 = nBase + lm;
    const int col1 = nBase + 16 + lm;
    const float bv0 = bias ? bias[col0] : 0.f;
    const float bv1 = bias ? bias[col1] : 0.f;
#pragma unroll
    for (int r = 0; r < 8; ++r) {
        int row = mTile + hi * 8 + r;   // C/D layout: lanes16-31 -> M = r+8
        if (row < M) {
            size_t rbase = (size_t)row * N;
            float v0 = act_apply(acc0[r] + bv0, act);
            float v1 = act_apply(acc1[r] + bv1, act);
            if (gate) {
                v0 *= (float)gate[rbase + col0];
                v1 *= (float)gate[rbase + col1];
            }
            C[rbase + col0] = (CT)v0;
            C[rbase + col1] = (CT)v1;
        }
    }
}

// ----- Edge gather + gated attention (one wave per (edge, head)) -----------
__global__ __launch_bounds__(256) void edge_gate_kernel(
    const __bf16* __restrict__ q, const __bf16* __restrict__ k,
    const __bf16* __restrict__ v, const __bf16* __restrict__ e,
    const int* __restrict__ src, const int* __restrict__ dst,
    const float* __restrict__ ln_g, const float* __restrict__ ln_b,
    __bf16* __restrict__ gate, __bf16* __restrict__ m0, int E)
{
    int wid  = blockIdx.x * 8 + (threadIdx.x >> 5);
    int lane = threadIdx.x & 31;
    if (wid >= E * 4) return;
    int edge = wid >> 2, h = wid & 3;
    int s = src[edge], d = dst[edge];
    const __bf16* kd = k + (size_t)d * 512 + h * 128;
    const __bf16* ks = k + (size_t)s * 512 + h * 128;
    const __bf16* qd = q + (size_t)d * 512 + h * 128;
    const __bf16* ee = e + (size_t)edge * 512 + h * 128;

    const float scale = rsqrtf(384.f);
    float alpha[12], sum = 0.f, sum2 = 0.f;
#pragma unroll
    for (int t = 0; t < 12; ++t) {
        int c = lane + 32 * t;
        float kv = (c < 128) ? (float)kd[c]
                 : (c < 256) ? (float)ks[c - 128]
                             : (float)ee[c - 256];
        float qv = (float)qd[c & 127];
        float av = qv * kv * scale;
        alpha[t] = av; sum += av; sum2 += av * av;
    }
#pragma unroll
    for (int off = 16; off; off >>= 1) {       // wave32 reduction
        sum  += __shfl_xor(sum,  off, 32);
        sum2 += __shfl_xor(sum2, off, 32);
    }
    float mean = sum * (1.f / 384.f);
    float var  = sum2 * (1.f / 384.f) - mean * mean;
    float rstd = rsqrtf(var + 1e-5f);

    const __bf16* vd = v + (size_t)d * 512 + h * 128;
    const __bf16* vs = v + (size_t)s * 512 + h * 128;
    size_t base = (size_t)wid * 384;
#pragma unroll
    for (int t = 0; t < 12; ++t) {
        int c = lane + 32 * t;
        float g = (alpha[t] - mean) * rstd * ln_g[c] + ln_b[c];
        g = 1.f / (1.f + __expf(-g));
        gate[base + c] = (__bf16)g;
        float mv = (c < 128) ? (float)vd[c]
                 : (c < 256) ? (float)vs[c - 128]
                             : (float)ee[c - 256];
        m0[base + c] = (__bf16)mv;
    }
}

// ----- LayerNorm over rows of 128 (one wave per row, in place) -------------
__global__ __launch_bounds__(256) void ln128_kernel(
    float* __restrict__ X, const float* __restrict__ g,
    const float* __restrict__ b, int R)
{
    int wid = blockIdx.x * 8 + (threadIdx.x >> 5);
    int lane = threadIdx.x & 31;
    if (wid >= R) return;
    float* row = X + (size_t)wid * 128;
    float v[4], s = 0.f, s2 = 0.f;
#pragma unroll
    for (int t = 0; t < 4; ++t) {
        v[t] = row[lane + 32 * t]; s += v[t]; s2 += v[t] * v[t];
    }
#pragma unroll
    for (int off = 16; off; off >>= 1) {
        s  += __shfl_xor(s,  off, 32);
        s2 += __shfl_xor(s2, off, 32);
    }
    float mean = s * (1.f / 128.f);
    float var  = s2 * (1.f / 128.f) - mean * mean;
    float rstd = rsqrtf(var + 1e-5f);
#pragma unroll
    for (int t = 0; t < 4; ++t) {
        int c = lane + 32 * t;
        row[c] = (v[t] - mean) * rstd * g[c] + b[c];
    }
}

// ----- segment_sum over dst via fp32 global atomics ------------------------
__global__ void scatter_add_kernel(const float* __restrict__ m5,
                                   const int* __restrict__ dst,
                                   float* __restrict__ agg, int E)
{
    int t = blockIdx.x * 256 + threadIdx.x;
    if (t >= E * 512) return;
    int edge = t >> 9, r = t & 511;
    atomicAdd(&agg[(size_t)dst[edge] * 512 + r], m5[t]);
}

// ----- BatchNorm (training-mode) stats + apply -----------------------------
__global__ __launch_bounds__(128) void bn_stats_kernel(
    const float* __restrict__ C, float* __restrict__ sums, int N)
{
    int c = threadIdx.x;
    int r0 = blockIdx.x * 64;
    int r1 = r0 + 64; if (r1 > N) r1 = N;
    float s = 0.f, s2 = 0.f;
    for (int r = r0; r < r1; ++r) {
        float v = C[(size_t)r * 128 + c];
        s += v; s2 += v * v;
    }
    atomicAdd(&sums[c], s);
    atomicAdd(&sums[128 + c], s2);
}

__global__ void bn_apply_kernel(const float* __restrict__ C,
                                const float* __restrict__ sums,
                                const float* __restrict__ g,
                                const float* __restrict__ b,
                                const float* __restrict__ skip,
                                __bf16* __restrict__ node_out,
                                float Ninv, int total)
{
    int t = blockIdx.x * 256 + threadIdx.x;
    if (t >= total) return;
    int c = t & 127;
    float mean = sums[c] * Ninv;
    float var  = sums[128 + c] * Ninv - mean * mean;
    float rstd = rsqrtf(var + 1e-5f);
    float v = (C[t] - mean) * rstd * g[c] + b[c];
    v = v / (1.f + __expf(-v));                 // SiLU
    node_out[t] = (__bf16)(v + skip[t]);
}

// ----- RBF expansion of bond lengths ---------------------------------------
__global__ void rbf_kernel(const float* __restrict__ ea,
                           __bf16* __restrict__ ef0, int E)
{
    int t = blockIdx.x * 256 + threadIdx.x;
    if (t >= E * 128) return;
    int e = t >> 7, bin = t & 127;
    float x0 = ea[e * 3], x1 = ea[e * 3 + 1], x2 = ea[e * 3 + 2];
    float d = sqrtf(x0 * x0 + x1 * x1 + x2 * x2);
    float c = 8.f * (float)bin / 127.f;
    const float gamma = 127.f / 8.f;
    float diff = d - c;
    ef0[t] = (__bf16)__expf(-gamma * diff * diff);
}

// ----- zero-pad K=92 -> 128 for WMMA alignment -----------------------------
__global__ void pad_x_kernel(const float* __restrict__ x,
                             __bf16* __restrict__ xp, int N)
{
    int t = blockIdx.x * 256 + threadIdx.x;
    if (t >= N * 128) return;
    int n = t >> 7, c = t & 127;
    xp[t] = (c < 92) ? (__bf16)x[n * 92 + c] : (__bf16)0.f;
}

// ----- mean pooling over graphs --------------------------------------------
__global__ void pool_kernel(const __bf16* __restrict__ node,
                            const int* __restrict__ batch,
                            float* __restrict__ pooled,
                            float* __restrict__ cnt, int N)
{
    int t = blockIdx.x * 256 + threadIdx.x;
    if (t >= N * 128) return;
    int n = t >> 7, c = t & 127;
    int g = batch[n];
    atomicAdd(&pooled[(size_t)g * 128 + c], (float)node[t]);
    if (c == 0) atomicAdd(&cnt[g], 1.f);
}

// ----- fc + out head (one block of 128 threads per graph) ------------------
__global__ __launch_bounds__(128) void head_kernel(
    const float* __restrict__ pooled, const float* __restrict__ cnt,
    const float* __restrict__ fcW, const float* __restrict__ fcb,
    const float* __restrict__ outW, const float* __restrict__ outb,
    float* __restrict__ out)
{
    int g = blockIdx.x, j = threadIdx.x;
    float inv = 1.f / fmaxf(cnt[g], 1.f);
    float acc = fcb[j];
    for (int c = 0; c < 128; ++c)
        acc += pooled[(size_t)g * 128 + c] * inv * fcW[c * 128 + j];
    acc = acc / (1.f + __expf(-acc));           // SiLU
    __shared__ float red[128];
    red[j] = acc * outW[j];
    __syncthreads();
    for (int s = 64; s; s >>= 1) {
        if (j < s) red[j] += red[j + s];
        __syncthreads();
    }
    if (j == 0) out[g] = red[0] + outb[0];
}

// ---------------------------------------------------------------------------
// Host side
// ---------------------------------------------------------------------------
static inline void* bump(char*& p, size_t bytes) {
    void* r = (void*)p;
    p += (bytes + 255) & ~(size_t)255;
    return r;
}

static inline dim3 ce(long n) { return dim3((unsigned)((n + 255) / 256)); }

// Pack W (fp32, [Ksrc,N] logical, padded to Kp rows) into packbuf, then GEMM.
template <typename AT, typename CT>
static void launch_gemm(const AT* A, const float* W, int Ksrc,
                        const float* bias, CT* C, const __bf16* gate,
                        int M, int N, int K, int act, __bf16* packbuf,
                        hipStream_t s) {
    pack_w_kernel<<<ce((long)K * N), 256, 0, s>>>(W, packbuf, Ksrc, K, N);
    dim3 grid((M + 63) / 64, N / 64);
    wmma_gemm_kernel<AT, CT><<<grid, 256, 0, s>>>(A, packbuf, bias, C, gate,
                                                  M, N, K, act);
}

extern "C" void kernel_launch(void* const* d_in, const int* in_sizes, int n_in,
                              void* d_out, int out_size, void* d_ws, size_t ws_size,
                              hipStream_t stream)
{
    const int N = in_sizes[0] / 92;
    const int E = in_sizes[1] / 3;
    const int L = 5;

    const float* x         = (const float*)d_in[0];
    const float* edge_attr = (const float*)d_in[1];
    const float* emb_W  = (const float*)d_in[2];
    const float* emb_b  = (const float*)d_in[3];
    const float* rbf_W1 = (const float*)d_in[4];
    const float* rbf_b1 = (const float*)d_in[5];
    const float* rbf_W2 = (const float*)d_in[6];
    const float* rbf_b2 = (const float*)d_in[7];
    const float* qW = (const float*)d_in[8];   const float* qb = (const float*)d_in[9];
    const float* kW = (const float*)d_in[10];  const float* kb = (const float*)d_in[11];
    const float* vW = (const float*)d_in[12];  const float* vb = (const float*)d_in[13];
    const float* eW = (const float*)d_in[14];  const float* eb = (const float*)d_in[15];
    const float* ln_g = (const float*)d_in[16]; const float* ln_b = (const float*)d_in[17];
    const float* muW1 = (const float*)d_in[18]; const float* mub1 = (const float*)d_in[19];
    const float* muW2 = (const float*)d_in[20]; const float* mub2 = (const float*)d_in[21];
    const float* mlW  = (const float*)d_in[22]; const float* mlb  = (const float*)d_in[23];
    const float* mln_g = (const float*)d_in[24]; const float* mln_b = (const float*)d_in[25];
    const float* cW = (const float*)d_in[26];  const float* cb = (const float*)d_in[27];
    const float* bn_g = (const float*)d_in[28]; const float* bn_b = (const float*)d_in[29];
    const float* sW = (const float*)d_in[30];  const float* sb = (const float*)d_in[31];
    const float* fc_W = (const float*)d_in[32]; const float* fc_b = (const float*)d_in[33];
    const float* out_W = (const float*)d_in[34]; const float* out_b = (const float*)d_in[35];
    const int* ei    = (const int*)d_in[36];
    const int* batch = (const int*)d_in[37];
    const int* src = ei;
    const int* dst = ei + E;

    // ---- workspace carve-up (bf16 for GEMM inputs except fp32 agg) ----------
    char* wp = (char*)d_ws;
    __bf16* xp    = (__bf16*)bump(wp, (size_t)N * 128 * 2);
    __bf16* nodeA = (__bf16*)bump(wp, (size_t)N * 128 * 2);
    __bf16* nodeB = (__bf16*)bump(wp, (size_t)N * 128 * 2);
    __bf16* packW = (__bf16*)bump(wp, (size_t)384 * 384 * 2);  // max K*N
    __bf16* ef0   = (__bf16*)bump(wp, (size_t)E * 128 * 2);
    __bf16* ef1   = (__bf16*)bump(wp, (size_t)E * 128 * 2);
    __bf16* ef    = (__bf16*)bump(wp, (size_t)E * 128 * 2);
    __bf16* qbuf  = (__bf16*)bump(wp, (size_t)N * 512 * 2);
    __bf16* kbuf  = (__bf16*)bump(wp, (size_t)N * 512 * 2);
    __bf16* vbuf  = (__bf16*)bump(wp, (size_t)N * 512 * 2);
    __bf16* ebuf  = (__bf16*)bump(wp, (size_t)E * 512 * 2);
    __bf16* gate  = (__bf16*)bump(wp, (size_t)E * 1536 * 2);
    __bf16* m0    = (__bf16*)bump(wp, (size_t)E * 1536 * 2);
    __bf16* m1    = (__bf16*)bump(wp, (size_t)E * 1536 * 2);
    float*  m4    = (float*) bump(wp, (size_t)E * 512 * 4);
    float*  agg   = (float*) bump(wp, (size_t)N * 512 * 4);
    float*  Cc    = (float*) bump(wp, (size_t)N * 128 * 4);
    float*  Sk    = (float*) bump(wp, (size_t)N * 128 * 4);
    float*  bnsum = (float*) bump(wp, (size_t)256 * 4);
    float*  pooled= (float*) bump(wp, (size_t)128 * 128 * 4);
    float*  cnt   = (float*) bump(wp, (size_t)128 * 4);

    // ---- embedding (K=92 zero-padded to 128 inside the pack kernel) -------
    pad_x_kernel<<<ce((long)N * 128), 256, 0, stream>>>(x, xp, N);
    launch_gemm<__bf16, __bf16>(xp, emb_W, 92, emb_b, nodeA, nullptr,
                                N, 128, 128, ACT_NONE, packW, stream);

    // ---- RBF edge features -------------------------------------------------
    rbf_kernel<<<ce((long)E * 128), 256, 0, stream>>>(edge_attr, ef0, E);
    launch_gemm<__bf16, __bf16>(ef0, rbf_W1, 128, rbf_b1, ef1, nullptr,
                                E, 128, 128, ACT_SOFTPLUS, packW, stream);
    launch_gemm<__bf16, __bf16>(ef1, rbf_W2, 128, rbf_b2, ef, nullptr,
                                E, 128, 128, ACT_NONE, packW, stream);

    __bf16* cur = nodeA;
    __bf16* nxt = nodeB;

    for (int i = 0; i < L; ++i) {
        const float* qWi = qW + (size_t)i * 128 * 512;
        const float* kWi = kW + (size_t)i * 128 * 512;
        const float* vWi = vW + (size_t)i * 128 * 512;
        const float* eWi = eW + (size_t)i * 128 * 512;
        const float* qbi = qb + (size_t)i * 512;
        const float* kbi = kb + (size_t)i * 512;
        const float* vbi = vb + (size_t)i * 512;
        const float* ebi = eb + (size_t)i * 512;
        const float* lngi = ln_g + (size_t)i * 384;
        const float* lnbi = ln_b + (size_t)i * 384;
        const float* muW1i = muW1 + (size_t)i * 384 * 384;
        const float* muW2i = muW2 + (size_t)i * 384 * 384;
        const float* mub1i = mub1 + (size_t)i * 384;
        const float* mub2i = mub2 + (size_t)i * 384;
        const float* mlWi = mlW + (size_t)i * 384 * 128;
        const float* mlbi = mlb + (size_t)i * 128;
        const float* mlngi = mln_g + (size_t)i * 128;
        const float* mlnbi = mln_b + (size_t)i * 128;
        const float* cWi = cW + (size_t)i * 512 * 128;
        const float* cbi = cb + (size_t)i * 128;
        const float* bngi = bn_g + (size_t)i * 128;
        const float* bnbi = bn_b + (size_t)i * 128;
        const float* sWi = sW + (size_t)i * 128 * 128;
        const float* sbi = sb + (size_t)i * 128;

        // projections (node -> q/k/v, ef -> e)
        launch_gemm<__bf16, __bf16>(cur, qWi, 128, qbi, qbuf, nullptr,
                                    N, 512, 128, ACT_NONE, packW, stream);
        launch_gemm<__bf16, __bf16>(cur, kWi, 128, kbi, kbuf, nullptr,
                                    N, 512, 128, ACT_NONE, packW, stream);
        launch_gemm<__bf16, __bf16>(cur, vWi, 128, vbi, vbuf, nullptr,
                                    N, 512, 128, ACT_NONE, packW, stream);
        launch_gemm<__bf16, __bf16>(ef, eWi, 128, ebi, ebuf, nullptr,
                                    E, 512, 128, ACT_NONE, packW, stream);

        // gather + alpha + LN(384) + sigmoid gate + message concat
        edge_gate_kernel<<<dim3((E * 4 + 7) / 8), 256, 0, stream>>>(
            qbuf, kbuf, vbuf, ebuf, src, dst, lngi, lnbi, gate, m0, E);

        // edge MLP: m1 = silu(m0 @ muW1 + b1); m0 <= (m1 @ muW2 + b2) * gate
        launch_gemm<__bf16, __bf16>(m0, muW1i, 384, mub1i, m1, nullptr,
                                    E * 4, 384, 384, ACT_SILU, packW, stream);
        launch_gemm<__bf16, __bf16>(m1, muW2i, 384, mub2i, m0, gate,
                                    E * 4, 384, 384, ACT_NONE, packW, stream);

        // m4 = m0 @ mlW + mlb ; per-(edge,head) LayerNorm over 128
        launch_gemm<__bf16, float>(m0, mlWi, 384, mlbi, m4, nullptr,
                                   E * 4, 128, 384, ACT_NONE, packW, stream);
        ln128_kernel<<<dim3((E * 4 + 7) / 8), 256, 0, stream>>>(m4, mlngi,
                                                                mlnbi, E * 4);

        // segment_sum over dst
        hipMemsetAsync(agg, 0, (size_t)N * 512 * 4, stream);
        scatter_add_kernel<<<ce((long)E * 512), 256, 0, stream>>>(m4, dst, agg, E);

        // out = silu(BN(agg @ cW + cb)) + node @ sW + sb
        launch_gemm<float, float>(agg, cWi, 512, cbi, Cc, nullptr,
                                  N, 128, 512, ACT_NONE, packW, stream);
        launch_gemm<__bf16, float>(cur, sWi, 128, sbi, Sk, nullptr,
                                   N, 128, 128, ACT_NONE, packW, stream);
        hipMemsetAsync(bnsum, 0, 256 * 4, stream);
        bn_stats_kernel<<<dim3((N + 63) / 64), 128, 0, stream>>>(Cc, bnsum, N);
        bn_apply_kernel<<<ce((long)N * 128), 256, 0, stream>>>(
            Cc, bnsum, bngi, bnbi, Sk, nxt, 1.f / (float)N, N * 128);

        __bf16* tmp = cur; cur = nxt; nxt = tmp;
    }

    // ---- mean pooling + head ----------------------------------------------
    hipMemsetAsync(pooled, 0, (size_t)128 * 128 * 4, stream);
    hipMemsetAsync(cnt, 0, 128 * 4, stream);
    pool_kernel<<<ce((long)N * 128), 256, 0, stream>>>(cur, batch, pooled, cnt, N);
    head_kernel<<<128, 128, 0, stream>>>(pooled, cnt, fc_W, fc_b, out_W, out_b,
                                         (float*)d_out);
}
